// NeuralRoughSimulator_29961691856930
// MI455X (gfx1250) — matrix operand order repaired and checked by reading, not compiled, not run on hardware
//
#include <hip/hip_runtime.h>

// ---------------------------------------------------------------------------
// Neural rough-volatility simulator for MI455X (gfx1250, wave32, WMMA).
// Phase 1: Wh = Z @ A.T as f16 WMMA GEMM (lower-triangular k-skip), fused
//          epilogue -> lv_base, written TIME-MAJOR [t][p].
// Phase 2: 256-step scan; each wave owns 32 paths (1/lane scalar state);
//          BOTH MLPs run fused per layer as v_wmma_f32_16x16x32_f16 chains,
//          weights staged in LDS (ds_load_b128 fragments), activations via
//          per-wave LDS staging, native v_tanh_f32 activations.
// Phase 3: tiled transpose [t][p] -> [p][t] output.
// ---------------------------------------------------------------------------

typedef __attribute__((ext_vector_type(16))) _Float16 v16h;
typedef __attribute__((ext_vector_type(8)))  float    v8f;

#define NPATH 16384
#define NSTEP 256

union FragH { v16h h; float4 q[2]; };
union V8FU  { v8f v; float e[8]; float4 q[2]; };

#define WMMA_F16(A,B,C) \
  __builtin_amdgcn_wmma_f32_16x16x32_f16(false,(A),false,(B),(short)0,(C),false,false)

// DS ops are in-order per wave; after this wait, cross-lane LDS data written
// by this wave is visible to its own loads (no workgroup barrier needed for
// per-wave-private staging buffers).
#define WAIT_DS() asm volatile("s_wait_dscnt 0" ::: "memory")

__device__ __forceinline__ float tanh_fast(float x){
#if __has_builtin(__builtin_amdgcn_tanhf)
  return __builtin_amdgcn_tanhf(x);          // v_tanh_f32 (CDNA5 TRANS op)
#else
  float r;
  asm volatile("v_tanh_f32 %0, %1\n\tv_nop" : "=v"(r) : "v"(x)); // +hazard slot
  return r;
#endif
}
__device__ __forceinline__ float sigmoid_fast(float x){
  return 1.0f / (1.0f + __expf(-x));
}

// A-fragment (16x32, this file's contiguous k order, same as loadB_rm): lanes 0-15 rows 0-15 hold K 0..15;
// lanes 16-31 hold K 16..31. Source is row-major [row][ks] halfs.
__device__ __forceinline__ FragH loadA_rm(const _Float16* base, int ks, int kc, int lane){
  const int row = lane & 15;
  const int kb  = (kc << 5) + ((lane & 16) ? 16 : 0);
  const _Float16* p = base + row * ks + kb;
  FragH f;
  f.q[0] = *(const float4*)(p);
  f.q[1] = *(const float4*)(p + 8);
  return f;
}

// B-fragment (32x16, contiguous k order, same as loadA_rm): lane(0-15)=column n, VGPR v packs k=2v,2v+1
// (lanes 16-31: k+16). Source row-major [n][ks] halfs -> one contiguous 32B.
__device__ __forceinline__ FragH loadB_rm(const _Float16* W, int ks, int kc, int nbase, int lane){
  const int n  = nbase + (lane & 15);
  const int kb = (kc << 5) + ((lane & 16) ? 16 : 0);
  const float4* s = (const float4*)(W + n * ks + kb);
  FragH f; f.q[0] = s[0]; f.q[1] = s[1];
  return f;
}

// D-tile (f32 16x16) epilogue: lane holds column f for 8 rows; apply
// bias+tanh, store f16 into [32][64] activation buffer.
__device__ __forceinline__ void store_act(_Float16* hb, int rowOff, int q,
                                          const v8f& c, float bias, int lane){
  V8FU u; u.v = c;
  const int f  = (q << 4) + (lane & 15);
  const int r0 = rowOff + ((lane & 16) ? 8 : 0);
#pragma unroll
  for (int r = 0; r < 8; ++r)
    hb[(r0 + r) * 64 + f] = (_Float16)tanh_fast(u.e[r] + bias);
}

// Both MLPs (15->64->64->64->1) fused, 32 paths, weights in LDS.
// Per-MLP weight blob (halfs): W0p[64][32] @0, W1p[64][64] @2048,
// W2p[64][64] @6144, W3p[16][64] @10240 (row 0 real).
__device__ __forceinline__ void run_mlps(const _Float16* wl,
    const float* bD0, const float* bD1, const float* bD2,
    const float* bF0, const float* bF1, const float* bF2,
    _Float16* hb0, _Float16* hb1,
    const FragH& x0, const FragH& x1, int lane,
    float& outD, float& outF){
  v8f z = {};
  // ---- layer 0 (K=32 padded), both MLPs ----
#pragma unroll
  for (int m = 0; m < 2; ++m){
    const _Float16* W = wl + m * 11264;
    const float*   bq = m ? bF0 : bD0;
    _Float16*      hb = m ? hb1 : hb0;
#pragma unroll
    for (int q = 0; q < 4; ++q){
      FragH b = loadB_rm(W, 32, 0, q << 4, lane);
      v8f c0 = WMMA_F16(x0.h, b.h, z);
      v8f c1 = WMMA_F16(x1.h, b.h, z);
      store_act(hb,  0, q, c0, bq[q], lane);
      store_act(hb, 16, q, c1, bq[q], lane);
    }
  }
  // ---- layers 1,2 (K=64), both MLPs ----
#pragma unroll
  for (int L = 0; L < 2; ++L){
    WAIT_DS();
    FragH a[2][4];
#pragma unroll
    for (int m = 0; m < 2; ++m){
      _Float16* hb = m ? hb1 : hb0;
      a[m][0] = loadA_rm(hb,           64, 0, lane);
      a[m][1] = loadA_rm(hb,           64, 1, lane);
      a[m][2] = loadA_rm(hb + 16 * 64, 64, 0, lane);
      a[m][3] = loadA_rm(hb + 16 * 64, 64, 1, lane);
    }
#pragma unroll
    for (int m = 0; m < 2; ++m){
      const _Float16* W  = wl + m * 11264 + (L ? 6144 : 2048);
      const float*   bq  = m ? (L ? bF2 : bF1) : (L ? bD2 : bD1);
      _Float16*      hb  = m ? hb1 : hb0;
#pragma unroll
      for (int q = 0; q < 4; ++q){
        FragH bb0 = loadB_rm(W, 64, 0, q << 4, lane);
        FragH bb1 = loadB_rm(W, 64, 1, q << 4, lane);
        v8f c0 = WMMA_F16(a[m][0].h, bb0.h, z); c0 = WMMA_F16(a[m][1].h, bb1.h, c0);
        v8f c1 = WMMA_F16(a[m][2].h, bb0.h, z); c1 = WMMA_F16(a[m][3].h, bb1.h, c1);
        store_act(hb,  0, q, c0, bq[q], lane);
        store_act(hb, 16, q, c1, bq[q], lane);
      }
    }
  }
  WAIT_DS();
  // ---- layer 3 (64 -> 1, N padded to 16), both MLPs ----
#pragma unroll
  for (int m = 0; m < 2; ++m){
    _Float16*       hb = m ? hb1 : hb0;
    const _Float16* W  = wl + m * 11264 + 10240;
    FragH f0 = loadA_rm(hb,           64, 0, lane);
    FragH f1 = loadA_rm(hb,           64, 1, lane);
    FragH f2 = loadA_rm(hb + 16 * 64, 64, 0, lane);
    FragH f3 = loadA_rm(hb + 16 * 64, 64, 1, lane);
    FragH b30 = loadB_rm(W, 64, 0, 0, lane);
    FragH b31 = loadB_rm(W, 64, 1, 0, lane);
    v8f c0 = WMMA_F16(f0.h, b30.h, z); c0 = WMMA_F16(f1.h, b31.h, c0);
    v8f c1 = WMMA_F16(f2.h, b30.h, z); c1 = WMMA_F16(f3.h, b31.h, c1);
    // Column n=0 lives in lanes 0 (rows 0-7) and 16 (rows 8-15) of each tile.
    // Scratch overlays the (now dead) activation buffer; DS in-order per wave
    // guarantees the A-fragment loads above read pre-store data.
    if ((lane & 15) == 0){
      V8FU u0; u0.v = c0; V8FU u1; u1.v = c1;
      float* o = (float*)hb + ((lane >> 4) << 3);
      *(float4*)(o)      = u0.q[0]; *(float4*)(o + 4)  = u0.q[1];
      *(float4*)(o + 16) = u1.q[0]; *(float4*)(o + 20) = u1.q[1];
    }
  }
  WAIT_DS();
  outD = ((float*)hb0)[lane];
  outF = ((float*)hb1)[lane];
}

// ---------------------------------------------------------------------------
// Prep kernels
// ---------------------------------------------------------------------------

// Volterra kernel A (f16, row-major [j][k]) + var_wh[j] = sum_k A^2.
__global__ __launch_bounds__(256) void prep_A(_Float16* __restrict__ Aw,
                                              float* __restrict__ varwh,
                                              const float* __restrict__ dt_p,
                                              const float* __restrict__ logH_p){
  const int j = blockIdx.x, k = threadIdx.x;
  const float H     = 0.5f / (1.0f + __expf(-logH_p[0]));
  const float dt    = dt_p[0];
  const float alpha = H + 0.5f;
  const float C     = sqrtf(2.0f * H) * powf(dt, H) / (H + 0.5f);
  float a = 0.0f;
  if (j >= k){
    float l = (float)(j - k);
    a = C * (powf(l + 1.0f, alpha) - powf(l, alpha));
  }
  Aw[j * 256 + k] = (_Float16)a;
  __shared__ float red[256];
  red[k] = a * a;
  __syncthreads();
  for (int s = 128; s > 0; s >>= 1){
    if (k < s) red[k] += red[k + s];
    __syncthreads();
  }
  if (k == 0) varwh[j] = red[0];
}

// dW -> Zh (f16, same layout, /sqrt(dt)) and dWT (f32, time-major) transpose.
__global__ __launch_bounds__(128) void prep_z(const float* __restrict__ dW,
                                              _Float16* __restrict__ Zh,
                                              float* __restrict__ dWT,
                                              const float* __restrict__ dt_p){
  __shared__ float tile[32][33];
  const float inv = 1.0f / sqrtf(dt_p[0]);
  const int pb = blockIdx.x * 32, tb = blockIdx.y * 32;
  const int tid = threadIdx.x;
#pragma unroll
  for (int i = 0; i < 8; ++i){
    int idx = tid + i * 128, r = idx >> 5, c = idx & 31;
    float v = dW[(pb + r) * NSTEP + tb + c];
    tile[r][c] = v;
    Zh[(pb + r) * NSTEP + tb + c] = (_Float16)(v * inv);
  }
  __syncthreads();
#pragma unroll
  for (int i = 0; i < 8; ++i){
    int idx = tid + i * 128, r = idx >> 5, c = idx & 31;
    dWT[(tb + r) * NPATH + pb + c] = tile[c][r];
  }
}

// Pad/convert both MLPs' weights to f16 blobs of 11264 halfs each.
__global__ __launch_bounds__(256) void prep_weights(_Float16* __restrict__ Wp,
    const float* __restrict__ dW0, const float* __restrict__ dW1,
    const float* __restrict__ dW2, const float* __restrict__ dW3,
    const float* __restrict__ fW0, const float* __restrict__ fW1,
    const float* __restrict__ fW2, const float* __restrict__ fW3){
  const int idx = blockIdx.x * 256 + threadIdx.x;
  if (idx >= 22528) return;
  const int m = idx / 11264, r = idx % 11264;
  const float* W0 = m ? fW0 : dW0;
  const float* W1 = m ? fW1 : dW1;
  const float* W2 = m ? fW2 : dW2;
  const float* W3 = m ? fW3 : dW3;
  float v;
  if (r < 2048){            // W0p [64][32] from [64][15]
    int n = r >> 5, k = r & 31;
    v = (k < 15) ? W0[n * 15 + k] : 0.0f;
  } else if (r < 6144){     // W1p [64][64]
    v = W1[r - 2048];
  } else if (r < 10240){    // W2p [64][64]
    v = W2[r - 6144];
  } else {                  // W3p [16][64] from [1][64]
    int q = r - 10240; int n = q >> 6, k = q & 63;
    v = (n == 0) ? W3[k] : 0.0f;
  }
  Wp[idx] = (_Float16)v;
}

// ---------------------------------------------------------------------------
// Phase 1: GEMM + fused lv_base epilogue (time-major output).
// ---------------------------------------------------------------------------
__global__ __launch_bounds__(128) void gemm_lvbase(
    const _Float16* __restrict__ Zh, const _Float16* __restrict__ Aw,
    const float* __restrict__ varwh, const float* __restrict__ init_var,
    float* __restrict__ lvT, const float* __restrict__ log_eta_p){
  const int wave = threadIdx.x >> 5, lane = threadIdx.x & 31;
  const int mtile = blockIdx.x * 4 + wave;      // 1024 m-tiles
  const int g     = blockIdx.y;                 // 4 n-groups of 64 cols
  const int mbase = mtile << 4;
  const float eta = __expf(log_eta_p[0]);
  v8f acc[4] = {};
  const int kcend = 2 * g + 2;                  // lower-triangular skip
  for (int kc = 0; kc < kcend; ++kc){
    FragH a = loadA_rm(Zh + mbase * 256, 256, kc, lane);
#pragma unroll
    for (int q = 0; q < 4; ++q){
      FragH b = loadB_rm(Aw, 256, kc, (g << 6) + (q << 4), lane);
      acc[q] = WMMA_F16(a.h, b.h, acc[q]);
    }
  }
  const int hiOff = (lane & 16) ? 8 : 0;
  float liv[8];
#pragma unroll
  for (int r = 0; r < 8; ++r)
    liv[r] = __logf(fmaxf(init_var[mbase + hiOff + r], 1e-10f));
#pragma unroll
  for (int q = 0; q < 4; ++q){
    const int   tcol = (g << 6) + (q << 4) + (lane & 15);
    const float bias = -0.5f * eta * eta * varwh[tcol];
    V8FU u; u.v = acc[q];
#pragma unroll
    for (int r = 0; r < 8; ++r)
      lvT[tcol * NPATH + mbase + hiOff + r] = liv[r] + eta * u.e[r] + bias;
  }
}

// ---------------------------------------------------------------------------
// Phase 2: sequential scan, 2 waves/block, 32 paths/wave, LDS-resident weights.
// Static LDS: 45056 (weights) + 16384 (h-buffers) + 4096 (x-buffers) = 64 KB.
// ---------------------------------------------------------------------------
__global__ __launch_bounds__(64) void scan_kernel(
    const float* __restrict__ lvT, float* __restrict__ wsT,
    const _Float16* __restrict__ Wp,
    const float* __restrict__ db0, const float* __restrict__ db1,
    const float* __restrict__ db2, const float* __restrict__ db3,
    const float* __restrict__ fb0, const float* __restrict__ fb1,
    const float* __restrict__ fb2, const float* __restrict__ fb3,
    const float* __restrict__ dt_p, const float* __restrict__ log_eta_p){
  __shared__ _Float16 wlds[22528];           // 45 KB: both MLPs' weights
  __shared__ _Float16 hbuf[2][2][32 * 64];   // 16 KB: [wave][mlp]
  __shared__ _Float16 xbuf[2][32 * 32];      //  4 KB: [wave]

  const int tid  = threadIdx.x;
  const int wave = tid >> 5;
  const int lane = tid & 31;
  const int l16  = lane & 15;
  const int p    = (blockIdx.x * 2 + wave) * 32 + lane;

  // Cooperative copy of the 45 KB weight blob into LDS (b128 stores).
  {
    const float4* src = (const float4*)Wp;
    float4*       dst = (float4*)wlds;
    for (int i = tid; i < 2816; i += 64) dst[i] = src[i];
  }
  __syncthreads();

  const float dt  = dt_p[0];
  const float eta = __expf(log_eta_p[0]);

  // Per-lane bias preload: feature f = 16q + (lane&15), fixed across steps.
  float bD0[4], bD1[4], bD2[4], bF0[4], bF1[4], bF2[4];
#pragma unroll
  for (int q = 0; q < 4; ++q){
    bD0[q] = db0[(q << 4) + l16]; bD1[q] = db1[(q << 4) + l16]; bD2[q] = db2[(q << 4) + l16];
    bF0[q] = fb0[(q << 4) + l16]; bF1[q] = fb1[(q << 4) + l16]; bF2[q] = fb2[(q << 4) + l16];
  }
  const float bD3 = db3[0], bF3 = fb3[0];

  float corr = 0.0f;
  float s1[2] = {0.0f, 0.0f};
  float s2[4] = {0.0f, 0.0f, 0.0f, 0.0f};
  float s3[8] = {0.0f, 0.0f, 0.0f, 0.0f, 0.0f, 0.0f, 0.0f, 0.0f};

  _Float16* xb  = xbuf[wave];
  _Float16* hb0 = hbuf[wave][0];
  _Float16* hb1 = hbuf[wave][1];

  for (int t = 0; t < NSTEP; ++t){
    const float dw  = wsT[t * NPATH + p];   // raw dW (time-major)
    const float lvb = lvT[t * NPATH + p];
    const float lv  = fminf(fmaxf(lvb + corr, -7.0f), 2.0f);
    const float nlv = (lv + 2.5f) * (1.0f / 2.25f);

    // Pack x = [s1, s2, s3, nlv, 0-pad] as 32 halfs into per-lane LDS row.
    union { float4 q[4]; _Float16 e[32]; } xv;
#pragma unroll
    for (int i = 0; i < 32; ++i) xv.e[i] = (_Float16)0.0f;
    xv.e[0] = (_Float16)s1[0]; xv.e[1] = (_Float16)s1[1];
#pragma unroll
    for (int i = 0; i < 4; ++i) xv.e[2 + i] = (_Float16)s2[i];
#pragma unroll
    for (int i = 0; i < 8; ++i) xv.e[6 + i] = (_Float16)s3[i];
    xv.e[14] = (_Float16)nlv;
    float4* xd = (float4*)(xb + lane * 32);
    xd[0] = xv.q[0]; xd[1] = xv.q[1]; xd[2] = xv.q[2]; xd[3] = xv.q[3];
    WAIT_DS();
    FragH x0 = loadA_rm(xb,           32, 0, lane);
    FragH x1 = loadA_rm(xb + 16 * 32, 32, 0, lane);

    float outD, outF;
    run_mlps(wlds, bD0, bD1, bD2, bF0, bF1, bF2, hb0, hb1, x0, x1, lane, outD, outF);

    const float drift    = 0.5f * tanh_fast(outD + bD3);
    const float diff     = 1.5f * sigmoid_fast(outF + bF3) + 0.1f;
    const float new_corr = corr + drift * dt + (diff - eta) * dw;
    const float lvn      = fminf(fmaxf(lvb + new_corr, -7.0f), 2.0f);
    wsT[t * NPATH + p] = __expf(lvn);       // overwrite consumed dW slot

    // Signature (level-3) update.
    const float dlv = lvn - lv;
    const float dxv[2] = {dt, dlv};
    const float dd[4]  = {dt * dt, dt * dlv, dlv * dt, dlv * dlv};
#pragma unroll
    for (int m = 0; m < 8; ++m)
      s3[m] += s2[m >> 1] * dxv[m & 1] + 0.5f * s1[m >> 2] * dd[m & 3]
             + (1.0f / 6.0f) * dxv[m >> 2] * dd[m & 3];
#pragma unroll
    for (int k = 0; k < 4; ++k)
      s2[k] += s1[k >> 1] * dxv[k & 1] + 0.5f * dd[k];
    s1[0] += dt; s1[1] += dlv;
    corr = new_corr;
  }
}

// ---------------------------------------------------------------------------
// Phase 3: [t][p] -> [p][t] output transpose (coalesced both sides).
// ---------------------------------------------------------------------------
__global__ __launch_bounds__(128) void transpose_out(const float* __restrict__ src,
                                                     float* __restrict__ dst){
  __shared__ float tile[32][33];
  const int pb = blockIdx.x * 32, tb = blockIdx.y * 32;
  const int tid = threadIdx.x;
#pragma unroll
  for (int i = 0; i < 8; ++i){
    int idx = tid + i * 128, r = idx >> 5, c = idx & 31;
    tile[r][c] = src[(tb + r) * NPATH + pb + c];
  }
  __syncthreads();
#pragma unroll
  for (int i = 0; i < 8; ++i){
    int idx = tid + i * 128, r = idx >> 5, c = idx & 31;
    dst[(pb + r) * NSTEP + tb + c] = tile[c][r];
  }
}

// ---------------------------------------------------------------------------
// Workspace layout (bytes): total ~40.2 MB.
// ---------------------------------------------------------------------------
#define OFF_ZH    0u            // f16 [16384][256] : 8,388,608
#define OFF_AW    8388608u      // f16 [256][256]   :   131,072
#define OFF_VW    8519680u      // f32 [256]        :     1,024
#define OFF_WP    8520704u      // f16 [22528]      :    45,056
#define OFF_LVT   8565760u      // f32 [256][16384] : 16,777,216
#define OFF_WST   25342976u     // f32 [256][16384] : 16,777,216 (dWT -> exp(lv))

extern "C" void kernel_launch(void* const* d_in, const int* in_sizes, int n_in,
                              void* d_out, int out_size, void* d_ws, size_t ws_size,
                              hipStream_t stream){
  const float* init_var = (const float*)d_in[0];
  const float* dW       = (const float*)d_in[1];
  const float* dt_p     = (const float*)d_in[2];
  const float* logH_p   = (const float*)d_in[3];
  const float* loge_p   = (const float*)d_in[4];
  const float* dW0 = (const float*)d_in[5],  *dW1 = (const float*)d_in[6];
  const float* dW2 = (const float*)d_in[7],  *dW3 = (const float*)d_in[8];
  const float* db0 = (const float*)d_in[9],  *db1 = (const float*)d_in[10];
  const float* db2 = (const float*)d_in[11], *db3 = (const float*)d_in[12];
  const float* fW0 = (const float*)d_in[13], *fW1 = (const float*)d_in[14];
  const float* fW2 = (const float*)d_in[15], *fW3 = (const float*)d_in[16];
  const float* fb0 = (const float*)d_in[17], *fb1 = (const float*)d_in[18];
  const float* fb2 = (const float*)d_in[19], *fb3 = (const float*)d_in[20];

  char* ws = (char*)d_ws;
  _Float16* Zh    = (_Float16*)(ws + OFF_ZH);
  _Float16* Aw    = (_Float16*)(ws + OFF_AW);
  float*    varwh = (float*)   (ws + OFF_VW);
  _Float16* Wp    = (_Float16*)(ws + OFF_WP);
  float*    lvT   = (float*)   (ws + OFF_LVT);
  float*    wsT   = (float*)   (ws + OFF_WST);

  prep_A      <<<256, 256, 0, stream>>>(Aw, varwh, dt_p, logH_p);
  prep_z      <<<dim3(NPATH / 32, NSTEP / 32), 128, 0, stream>>>(dW, Zh, wsT, dt_p);
  prep_weights<<<88, 256, 0, stream>>>(Wp, dW0, dW1, dW2, dW3, fW0, fW1, fW2, fW3);
  gemm_lvbase <<<dim3(256, 4), 128, 0, stream>>>(Zh, Aw, varwh, init_var, lvT, loge_p);
  scan_kernel <<<256, 64, 0, stream>>>(lvT, wsT, Wp,
                                       db0, db1, db2, db3, fb0, fb1, fb2, fb3,
                                       dt_p, loge_p);
  transpose_out<<<dim3(NPATH / 32, NSTEP / 32), 128, 0, stream>>>(wsT, (float*)d_out);
}
